// TensorNetRepresentation_13383118094948
// MI455X (gfx1250) — matrix operand order repaired and checked by reading, not compiled, not run on hardware
//
#include <hip/hip_runtime.h>

#define H_DIM 64
#define R_DIM 32

typedef __attribute__((ext_vector_type(16))) _Float16 v16h;
typedef __attribute__((ext_vector_type(8)))  float    v8f;

// ---------------------------------------------------------------- WMMA core
__device__ __forceinline__ v8f wmma_f16(v16h a, v16h b, v8f c) {
  // D = A(16x32 f16) * B(32x16 f16) + C(16x16 f32)
  return __builtin_amdgcn_wmma_f32_16x16x32_f16(false, a, false, b, (short)0, c,
                                                false, false);
}

// A-matrix gather from row-major LDS tile (16 x strideH halves).
// A layout: lane L holds row M=L%16; K(lane,v,p) =
// (L>=16?16:0) + 2*v + p   (+ kbase for the K-chunk)
__device__ __forceinline__ v16h lds_gather_A(const _Float16* As, int strideH,
                                             int lane, int kbase) {
  union { v16h v; unsigned u[8]; } r;
  const _Float16* row = As + (size_t)(lane & 15) * strideH;
  int koff = kbase + ((lane >> 4) << 4);
#pragma unroll
  for (int i = 0; i < 8; ++i) {
    int k = koff + (i << 1);   // pairs p=0,1 contiguous
    r.u[i] = *(const unsigned*)(row + k);
  }
  return r.v;
}

// B-matrix gather from column-major LDS staging Bs[n][strideK] (halves).
// B layout: lane L holds column N=L%16; lanes 16..31 hold K+16 half.
__device__ __forceinline__ v16h lds_gather_B(const _Float16* Bs, int strideK,
                                             int lane, int kbase) {
  union { v16h v; unsigned u[8]; } r;
  const _Float16* col = Bs + (size_t)(lane & 15) * strideK;
  int koff = kbase + ((lane >> 4) << 4);
#pragma unroll
  for (int i = 0; i < 8; ++i) r.u[i] = *(const unsigned*)(col + koff + (i << 1));
  return r.v;
}

// Pre-swizzled global tile (512 halves = one 32x16 B tile or 16x32 A tile):
// lane's 16 halves are contiguous -> two global_load_b128 per operand.
__device__ __forceinline__ v16h load_tile(const _Float16* __restrict__ w,
                                          int tile, int lane) {
  return *(const v16h*)(w + (size_t)tile * 512 + lane * 16);
}

__device__ __forceinline__ void atomAddF(float* p, float v) {
  unsafeAtomicAdd(p, v);   // native global_atomic_add_f32 (no-return)
}

// Async DMA: copy 128 B from global to LDS without touching VGPR data.
// IOFFSET is added to BOTH the LDS and global address (ISA 08_async_tensor §4.4),
// so one base pair serves all 8 b128 beats. Tracked by ASYNCcnt.
__device__ __forceinline__ void async_copy_row128(void* lds_dst,
                                                  const void* gsrc) {
  unsigned ldsa = (unsigned)(uintptr_t)lds_dst;          // LDS aperture: low 32b
  unsigned long long ga = (unsigned long long)(uintptr_t)gsrc;
  asm volatile(
      "global_load_async_to_lds_b128 %0, %1, off offset:0\n\t"
      "global_load_async_to_lds_b128 %0, %1, off offset:16\n\t"
      "global_load_async_to_lds_b128 %0, %1, off offset:32\n\t"
      "global_load_async_to_lds_b128 %0, %1, off offset:48\n\t"
      "global_load_async_to_lds_b128 %0, %1, off offset:64\n\t"
      "global_load_async_to_lds_b128 %0, %1, off offset:80\n\t"
      "global_load_async_to_lds_b128 %0, %1, off offset:96\n\t"
      "global_load_async_to_lds_b128 %0, %1, off offset:112"
      :: "v"(ldsa), "v"(ga) : "memory");
}

__device__ __forceinline__ void wait_async0() {
  asm volatile("s_wait_asynccnt 0" ::: "memory");
}

// ---------------------------------------------------------------- prep kernels
__global__ void k_zero(float* __restrict__ p, int n4) {
  int i = blockIdx.x * blockDim.x + threadIdx.x;
  if (i < n4) ((float4*)p)[i] = make_float4(0.f, 0.f, 0.f, 0.f);
}

__global__ void k_embed(const int* __restrict__ z, const float* __restrict__ emb,
                        _Float16* __restrict__ zi, int n) {
  int i = blockIdx.x * blockDim.x + threadIdx.x;
  if (i < n) zi[i] = (_Float16)emb[(size_t)z[i >> 6] * H_DIM + (i & 63)];
}

// B operand swizzle for X @ W^T : B[k][n] = W[n][k], W row-major (Ntot x Ktot)
__global__ void k_swizzle_B(const float* __restrict__ W, _Float16* __restrict__ dst,
                            int Ktot, int Ntot) {
  int total = (Ktot >> 5) * (Ntot >> 4) * 512;
  int i = blockIdx.x * blockDim.x + threadIdx.x;
  if (i >= total) return;
  int t = i >> 9, w = i & 511, lane = w >> 4, h = w & 15;
  int ntTiles = Ntot >> 4;
  int kt = t / ntTiles, nt = t - kt * ntTiles;
  int n = nt * 16 + (lane & 15);
  int k = kt * 32 + ((lane >> 4) << 4) + h;
  dst[i] = (_Float16)W[(size_t)n * Ktot + k];
}

// A operand swizzle for W @ X : A[m][k] = W[m][k], W row-major (Mtot x Ktot)
__global__ void k_swizzle_A(const float* __restrict__ W, _Float16* __restrict__ dst,
                            int Ktot, int Mtot) {
  int total = (Mtot >> 4) * (Ktot >> 5) * 512;
  int i = blockIdx.x * blockDim.x + threadIdx.x;
  if (i >= total) return;
  int t = i >> 9, w = i & 511, lane = w >> 4, h = w & 15;
  int ktTiles = Ktot >> 5;
  int mt = t / ktTiles, kt = t - mt * ktTiles;
  int m = mt * 16 + (lane & 15);
  int vv = h >> 1, p = h & 1;
  int k = kt * 32 + ((lane >> 4) << 4) + (vv << 1) + p;
  dst[i] = (_Float16)W[(size_t)m * Ktot + k];
}

// ---------------------------------------------------------------- pair kernel
// One wave32 per 16 pairs. 28 WMMAs per tile + 320 f32 atomics per lane.
__global__ __launch_bounds__(32) void k_pairs(
    const int* __restrict__ pairs, const float* __restrict__ d_ij,
    const float* __restrict__ r_ij, const _Float16* __restrict__ zi,
    const _Float16* __restrict__ w_emb2, const float* __restrict__ emb2_b,
    const _Float16* __restrict__ w_dp1, const float* __restrict__ dp1_b,
    const _Float16* __restrict__ w_dp2, const float* __restrict__ dp2_b,
    const _Float16* __restrict__ w_dp3, const float* __restrict__ dp3_b,
    float* __restrict__ accum, int P) {
  __shared__ _Float16 zijA[16 * 128];   // A tile for Zij GEMM (16 pairs x 2H)
  __shared__ _Float16 rbfA[16 * 32];    // A tile for rbf GEMMs
  __shared__ float sh_rc[16], sh_ux[16], sh_uy[16], sh_uz[16];
  __shared__ int sh_src[16], sh_dst[16];

  const int lane = threadIdx.x;
  const int pbase = blockIdx.x * 16;

  // prefetch next tile's geometry (global_prefetch_b8)
  __builtin_prefetch(d_ij + pbase + 16, 0, 0);
  __builtin_prefetch(r_ij + 3 * (pbase + 16), 0, 0);

  if (lane < 16) {
    int p = pbase + lane;
    int s = pairs[p], dd = pairs[P + p];
    float d = d_ij[p];
    float rx = r_ij[3 * p], ry = r_ij[3 * p + 1], rz = r_ij[3 * p + 2];
    float inv = 1.0f / d;
    float rc = (d < 0.5f) ? 0.5f * (__cosf(6.2831853071795864f * d) + 1.0f) : 0.0f;
    sh_src[lane] = s; sh_dst[lane] = dd;
    sh_rc[lane] = rc;
    sh_ux[lane] = rx * inv; sh_uy[lane] = ry * inv; sh_uz[lane] = rz * inv;
    const float start = 0.60653065971263342f;           // exp(-CUTOFF)
    const float step = (1.0f - start) * (1.0f / 31.0f);
    float tb = 0.0625f * (1.0f - start);
    const float beta = 1.0f / (tb * tb);
    float e = __expf(-10.0f * d);                       // alpha = 5/CUTOFF
#pragma unroll
    for (int k = 0; k < 32; ++k) {
      float t = e - (start + step * (float)k);
      rbfA[lane * 32 + k] = (_Float16)(__expf(-beta * t * t) * rc);
    }
  }
  __syncthreads();
  {  // async DMA gather zi[src]|zi[dst] rows straight into the LDS A tile
    int m = lane & 15, hh = lane >> 4;
    int a = hh ? sh_dst[m] : sh_src[m];
    async_copy_row128(zijA + m * 128 + hh * 64, zi + (size_t)a * H_DIM);
  }
  wait_async0();
  __syncthreads();

  // Hoist A-operand gathers: invariant across the 4 N-tiles.
  v16h aZ0 = lds_gather_A(zijA, 128, lane, 0);
  v16h aZ1 = lds_gather_A(zijA, 128, lane, 32);
  v16h aZ2 = lds_gather_A(zijA, 128, lane, 64);
  v16h aZ3 = lds_gather_A(zijA, 128, lane, 96);
  v16h ar  = lds_gather_A(rbfA, 32, lane, 0);

#pragma unroll
  for (int nt = 0; nt < 4; ++nt) {
    v8f accZ = {};
    accZ = wmma_f16(aZ0, load_tile(w_emb2, 0 * 4 + nt, lane), accZ);
    accZ = wmma_f16(aZ1, load_tile(w_emb2, 1 * 4 + nt, lane), accZ);
    accZ = wmma_f16(aZ2, load_tile(w_emb2, 2 * 4 + nt, lane), accZ);
    accZ = wmma_f16(aZ3, load_tile(w_emb2, 3 * 4 + nt, lane), accZ);
    v8f a1 = {}, a2 = {}, a3 = {};
    a1 = wmma_f16(ar, load_tile(w_dp1, nt, lane), a1);
    a2 = wmma_f16(ar, load_tile(w_dp2, nt, lane), a2);
    a3 = wmma_f16(ar, load_tile(w_dp3, nt, lane), a3);

    int n = nt * 16 + (lane & 15);
    float b0 = emb2_b[n], b1 = dp1_b[n], b2 = dp2_b[n], b3 = dp3_b[n];
#pragma unroll
    for (int r = 0; r < 8; ++r) {
      int m = r + ((lane >> 4) << 3);                 // D row (pair in tile)
      float rc = sh_rc[m];
      float C = rc * (accZ[r] + b0);
      float f1 = (a1[r] + b1) * C;
      float f2 = (a2[r] + b2) * C * 10.0f;
      float f3 = (a3[r] + b3) * C * 100.0f;
      float ux = sh_ux[m], uy = sh_uy[m], uz = sh_uz[m];
      float tr3 = (ux * ux + uy * uy + uz * uz) * (1.0f / 3.0f);
      float* base = accum + ((size_t)sh_src[m] * H_DIM + n) * 10;
      atomAddF(base + 0, f1);
      atomAddF(base + 1, f2 * ux);
      atomAddF(base + 2, f2 * uy);
      atomAddF(base + 3, f2 * uz);
      atomAddF(base + 4, f3 * (ux * ux - tr3));
      atomAddF(base + 5, f3 * (uy * uy - tr3));
      atomAddF(base + 6, f3 * (uz * uz - tr3));
      atomAddF(base + 7, f3 * ux * uy);
      atomAddF(base + 8, f3 * ux * uz);
      atomAddF(base + 9, f3 * uy * uz);
    }
  }
}

// ---------------------------------------------------------------- atom kernel
// One wave32 per 16 atoms: tn + layernorm + MLP (64 WMMAs) + lt mixing (80 WMMAs)
__global__ __launch_bounds__(32) void k_atoms(
    const float* __restrict__ accum,
    const float* __restrict__ ln_g, const float* __restrict__ ln_b,
    const _Float16* __restrict__ w_ls1, const float* __restrict__ ls1_b,
    const _Float16* __restrict__ w_ls2, const float* __restrict__ ls2_b,
    const _Float16* __restrict__ w_lt0, const _Float16* __restrict__ w_lt1,
    const _Float16* __restrict__ w_lt2, float* __restrict__ out) {
  __shared__ float    tn[16 * 64];        // 4 KB
  __shared__ _Float16 nA[16 * 64];        // 2 KB  (layernorm out, A staging)
  __shared__ _Float16 A2[16 * 128];       // 4 KB  (MLP hidden, A staging)
  __shared__ float    n3[16 * 192];       // 12 KB (MLP out)
  __shared__ _Float16 Bs[10][16 * 64];    // 20 KB (comp B staging, col-major)

  const int lane = threadIdx.x;
  const int abase = blockIdx.x * 16;

  // ||I+A+S||_F^2 = 3 f1^2 + 2|a|^2 + (s_d^2 + 2 s_o^2); also stage comps as B
  for (int idx = lane; idx < 16 * 64; idx += 32) {
    int aL = idx >> 6, ch = idx & 63;
    const float* c = accum + ((size_t)(abase + aL) * H_DIM + ch) * 10;
    float c0 = c[0], c1 = c[1], c2 = c[2], c3 = c[3], c4 = c[4];
    float c5 = c[5], c6 = c[6], c7 = c[7], c8 = c[8], c9 = c[9];
    tn[idx] = 3.f * c0 * c0 + 2.f * (c1 * c1 + c2 * c2 + c3 * c3) +
              c4 * c4 + c5 * c5 + c6 * c6 + 2.f * (c7 * c7 + c8 * c8 + c9 * c9);
    int bi = aL * 64 + ch;   // column-major: [atom][k=channel]
    Bs[0][bi] = (_Float16)c0; Bs[1][bi] = (_Float16)c1; Bs[2][bi] = (_Float16)c2;
    Bs[3][bi] = (_Float16)c3; Bs[4][bi] = (_Float16)c4; Bs[5][bi] = (_Float16)c5;
    Bs[6][bi] = (_Float16)c6; Bs[7][bi] = (_Float16)c7; Bs[8][bi] = (_Float16)c8;
    Bs[9][bi] = (_Float16)c9;
  }
  __syncthreads();

  if (lane < 16) {  // layernorm over H per atom
    const float* t = tn + lane * 64;
    float s = 0.f, s2 = 0.f;
#pragma unroll
    for (int ch = 0; ch < 64; ++ch) { float x = t[ch]; s += x; s2 += x * x; }
    float mu = s * (1.f / 64.f);
    float var = s2 * (1.f / 64.f) - mu * mu;
    float rst = rsqrtf(var + 1e-5f);
#pragma unroll
    for (int ch = 0; ch < 64; ++ch)
      nA[lane * 64 + ch] = (_Float16)((t[ch] - mu) * rst * ln_g[ch] + ln_b[ch]);
  }
  __syncthreads();

  // MLP layer 1: (16x64) @ (64x128) -> silu
  {
    v16h aN0 = lds_gather_A(nA, 64, lane, 0);
    v16h aN1 = lds_gather_A(nA, 64, lane, 32);
#pragma unroll
    for (int nt = 0; nt < 8; ++nt) {
      v8f acc = {};
      acc = wmma_f16(aN0, load_tile(w_ls1, nt, lane), acc);
      acc = wmma_f16(aN1, load_tile(w_ls1, 8 + nt, lane), acc);
      int n = nt * 16 + (lane & 15);
      float b = ls1_b[n];
#pragma unroll
      for (int r = 0; r < 8; ++r) {
        int m = r + ((lane >> 4) << 3);
        float x = acc[r] + b;
        A2[m * 128 + n] = (_Float16)(x / (1.f + __expf(-x)));
      }
    }
  }
  __syncthreads();

  // MLP layer 2: (16x128) @ (128x192) -> silu -> n3
  {
    v16h aH0 = lds_gather_A(A2, 128, lane, 0);
    v16h aH1 = lds_gather_A(A2, 128, lane, 32);
    v16h aH2 = lds_gather_A(A2, 128, lane, 64);
    v16h aH3 = lds_gather_A(A2, 128, lane, 96);
#pragma unroll
    for (int nt = 0; nt < 12; ++nt) {
      v8f acc = {};
      acc = wmma_f16(aH0, load_tile(w_ls2, 0 * 12 + nt, lane), acc);
      acc = wmma_f16(aH1, load_tile(w_ls2, 1 * 12 + nt, lane), acc);
      acc = wmma_f16(aH2, load_tile(w_ls2, 2 * 12 + nt, lane), acc);
      acc = wmma_f16(aH3, load_tile(w_ls2, 3 * 12 + nt, lane), acc);
      int n = nt * 16 + (lane & 15);
      float b = ls2_b[n];
#pragma unroll
      for (int r = 0; r < 8; ++r) {
        int m = r + ((lane >> 4) << 3);
        float x = acc[r] + b;
        n3[m * 192 + n] = x / (1.f + __expf(-x));
      }
    }
  }
  __syncthreads();

  // Channel mixing: out_c[g][atom] = lt_sel[g,h] @ comp_c[atom][h], then scale
#pragma unroll
  for (int mt = 0; mt < 4; ++mt) {
    float ro[10][8];
#pragma unroll
    for (int c = 0; c < 10; ++c) {
      const _Float16* wA = (c == 0) ? w_lt0 : (c < 4 ? w_lt1 : w_lt2);
      v8f acc = {};
      acc = wmma_f16(load_tile(wA, mt * 2 + 0, lane),
                     lds_gather_B(Bs[c], 64, lane, 0), acc);
      acc = wmma_f16(load_tile(wA, mt * 2 + 1, lane),
                     lds_gather_B(Bs[c], 64, lane, 32), acc);
#pragma unroll
      for (int r = 0; r < 8; ++r) ro[c][r] = acc[r];
    }
    int aL = lane & 15;
    size_t atomG = (size_t)abase + aL;
#pragma unroll
    for (int r = 0; r < 8; ++r) {
      int g = mt * 16 + r + ((lane >> 4) << 3);
      float nI = n3[aL * 192 + g * 3 + 0];
      float nAv = n3[aL * 192 + g * 3 + 1];
      float nS = n3[aL * 192 + g * 3 + 2];
      float i0 = ro[0][r] * nI;
      float ax = ro[1][r] * nAv, ay = ro[2][r] * nAv, az = ro[3][r] * nAv;
      float sxx = ro[4][r] * nS, syy = ro[5][r] * nS, szz = ro[6][r] * nS;
      float sxy = ro[7][r] * nS, sxz = ro[8][r] * nS, syz = ro[9][r] * nS;
      float* o = out + (atomG * H_DIM + g) * 9;
      o[0] = i0 + sxx; o[1] = -az + sxy; o[2] =  ay + sxz;
      o[3] =  az + sxy; o[4] = i0 + syy; o[5] = -ax + syz;
      o[6] = -ay + sxz; o[7] =  ax + syz; o[8] = i0 + szz;
    }
  }
}

// ---------------------------------------------------------------- launcher
extern "C" void kernel_launch(void* const* d_in, const int* in_sizes, int n_in,
                              void* d_out, int out_size, void* d_ws, size_t ws_size,
                              hipStream_t stream) {
  (void)n_in; (void)out_size; (void)ws_size;
  const int*   z      = (const int*)  d_in[0];
  const int*   pairs  = (const int*)  d_in[1];
  const float* dij    = (const float*)d_in[2];
  const float* rij    = (const float*)d_in[3];
  const float* emb    = (const float*)d_in[4];
  const float* emb2_w = (const float*)d_in[5];
  const float* emb2_b = (const float*)d_in[6];
  const float* dp1_w  = (const float*)d_in[7];
  const float* dp1_b  = (const float*)d_in[8];
  const float* dp2_w  = (const float*)d_in[9];
  const float* dp2_b  = (const float*)d_in[10];
  const float* dp3_w  = (const float*)d_in[11];
  const float* dp3_b  = (const float*)d_in[12];
  const float* lt0_w  = (const float*)d_in[13];
  const float* lt1_w  = (const float*)d_in[14];
  const float* lt2_w  = (const float*)d_in[15];
  const float* ls1_w  = (const float*)d_in[16];
  const float* ls1_b  = (const float*)d_in[17];
  const float* ls2_w  = (const float*)d_in[18];
  const float* ls2_b  = (const float*)d_in[19];
  const float* ln_g   = (const float*)d_in[20];
  const float* ln_b   = (const float*)d_in[21];

  const int NA = in_sizes[0];       // 10000
  const int P  = in_sizes[2];       // 100000

  char* ws = (char*)d_ws;
  size_t off = 0;
  auto carve = [&](size_t bytes) -> void* {
    void* p = ws + off;
    off = (off + bytes + 255) & ~(size_t)255;
    return p;
  };
  _Float16* zi    = (_Float16*)carve((size_t)NA * H_DIM * 2);
  _Float16* wEmb2 = (_Float16*)carve(16 * 512 * 2);
  _Float16* wDp1  = (_Float16*)carve(4 * 512 * 2);
  _Float16* wDp2  = (_Float16*)carve(4 * 512 * 2);
  _Float16* wDp3  = (_Float16*)carve(4 * 512 * 2);
  _Float16* wLs1  = (_Float16*)carve(16 * 512 * 2);
  _Float16* wLs2  = (_Float16*)carve(48 * 512 * 2);
  _Float16* wLt0  = (_Float16*)carve(8 * 512 * 2);
  _Float16* wLt1  = (_Float16*)carve(8 * 512 * 2);
  _Float16* wLt2  = (_Float16*)carve(8 * 512 * 2);
  float*    accum = (float*)carve((size_t)NA * H_DIM * 10 * 4);

  int accN4 = NA * H_DIM * 10 / 4;
  k_zero<<<(accN4 + 255) / 256, 256, 0, stream>>>(accum, accN4);
  k_embed<<<(NA * H_DIM + 255) / 256, 256, 0, stream>>>(z, emb, zi, NA * H_DIM);

  auto swB = [&](const float* W, _Float16* dst, int K, int N) {
    int tot = (K / 32) * (N / 16) * 512;
    k_swizzle_B<<<(tot + 255) / 256, 256, 0, stream>>>(W, dst, K, N);
  };
  swB(emb2_w, wEmb2, 128, 64);
  swB(dp1_w, wDp1, 32, 64);
  swB(dp2_w, wDp2, 32, 64);
  swB(dp3_w, wDp3, 32, 64);
  swB(ls1_w, wLs1, 64, 128);
  swB(ls2_w, wLs2, 128, 192);

  auto swA = [&](const float* W, _Float16* dst) {
    int tot = 4 * 2 * 512;
    k_swizzle_A<<<(tot + 255) / 256, 256, 0, stream>>>(W, dst, 64, 64);
  };
  swA(lt0_w, wLt0);
  swA(lt1_w, wLt1);
  swA(lt2_w, wLt2);

  k_pairs<<<P / 16, 32, 0, stream>>>(pairs, dij, rij, zi, wEmb2, emb2_b,
                                     wDp1, dp1_b, wDp2, dp2_b, wDp3, dp3_b,
                                     accum, P);
  k_atoms<<<NA / 16, 32, 0, stream>>>(accum, ln_g, ln_b, wLs1, ls1_b, wLs2,
                                      ls2_b, wLt0, wLt1, wLt2, (float*)d_out);
}
